// Model_39633958207890
// MI455X (gfx1250) — matrix operand order repaired and checked by reading, not compiled, not run on hardware
//
#include <hip/hip_runtime.h>
#include <hip/hip_bf16.h>

// ---------------------------------------------------------------------------
// SparseVggNet forward for MI455X (gfx1250, wave32, WMMA).
// - activations: bf16 NHWC, channel stride rounded to 16 (only C=3 padded)
// - weights: pre-packed on device into WMMA A-fragment lane order (bf16)
// - convs: implicit GEMM on v_wmma_f32_16x16x32_bf16, branch-free inner loop
//   (invalid taps: clamped address + dword AND-mask instead of exec branches)
// - sparse BN: LDS ds_add_f32 reduction + fused BN/ReLU/mask apply
// ---------------------------------------------------------------------------

typedef __attribute__((ext_vector_type(16))) __bf16    bf16x16;
typedef __attribute__((ext_vector_type(8)))  float     v8f;
typedef __attribute__((ext_vector_type(8)))  unsigned  u32x8;

#define NEGBIG -1e30f
#define BN_EPS 1e-4f

__device__ __forceinline__ int kmapA(int e, int half) {
  // A-fragment (16-bit, 16x32), contiguous order: element e of lane-half -> K = 16*half + e
  return (half << 4) + e;
}

// ---------------------------------------------------------------------------
// weight packing: OIHW f32 -> [mt][tap][kc][lane][e] bf16 (A-fragment order)
// ---------------------------------------------------------------------------
__global__ void pack_conv_kernel(const float* __restrict__ w, __bf16* __restrict__ wp,
                                 int cin, int cout) {
  int nkc = (cin + 31) >> 5;
  long long total = ((long long)(cout >> 4) * 9 * nkc) << 9;
  long long idx = (long long)blockIdx.x * blockDim.x + threadIdx.x;
  if (idx >= total) return;
  int e = (int)(idx & 15);
  int lane = (int)((idx >> 4) & 31);
  long long g = idx >> 9;
  int kc = (int)(g % nkc); g /= nkc;
  int t = (int)(g % 9);
  int mt = (int)(g / 9);
  int half = lane >> 4, n = lane & 15;
  int c = (kc << 5) + kmapA(e, half);
  int m = mt * 16 + n;
  wp[idx] = (c < cin) ? (__bf16)w[((long long)m * cin + c) * 9 + t] : (__bf16)0.0f;
}

// w_final (128,96,3,3) -> [mt(8)][kc(27)][lane][e], patch k = (y*3+x)*96 + c
__global__ void pack_final_kernel(const float* __restrict__ wf, __bf16* __restrict__ wp) {
  long long total = (8LL * 27) << 9;
  long long idx = (long long)blockIdx.x * blockDim.x + threadIdx.x;
  if (idx >= total) return;
  int e = (int)(idx & 15);
  int lane = (int)((idx >> 4) & 31);
  long long g = idx >> 9;
  int kc = (int)(g % 27);
  int mt = (int)(g / 27);
  int half = lane >> 4, n = lane & 15;
  int k = (kc << 5) + kmapA(e, half);
  int c = k % 96, tap = k / 96;
  int m = mt * 16 + n;
  wp[idx] = (__bf16)wf[((long long)m * 96 + c) * 9 + tap];
}

// Wlin (3755,128) -> [mt(235)][kc(4)][lane][e], rows >= 3755 zero
__global__ void pack_linear_kernel(const float* __restrict__ Wl, __bf16* __restrict__ wp) {
  long long total = (235LL * 4) << 9;
  long long idx = (long long)blockIdx.x * blockDim.x + threadIdx.x;
  if (idx >= total) return;
  int e = (int)(idx & 15);
  int lane = (int)((idx >> 4) & 31);
  long long g = idx >> 9;
  int kc = (int)(g & 3);
  int mt = (int)(g >> 2);
  int half = lane >> 4, n = lane & 15;
  int k = (kc << 5) + kmapA(e, half);
  int m = mt * 16 + n;
  wp[idx] = (m < 3755) ? (__bf16)Wl[(long long)m * 128 + k] : (__bf16)0.0f;
}

// ---------------------------------------------------------------------------
// prep: h = x * m  (NCHW f32 -> NHWC bf16, C=3 padded to stride 16)
// ---------------------------------------------------------------------------
__global__ void prep_kernel(const float* __restrict__ x, const int* __restrict__ mask,
                            __bf16* __restrict__ act, float* __restrict__ mf,
                            int B, int H, int W) {
  long long HW = (long long)H * W;
  long long total = (long long)B * HW * 16;
  for (long long idx = (long long)blockIdx.x * blockDim.x + threadIdx.x; idx < total;
       idx += (long long)gridDim.x * blockDim.x) {
    int c = (int)(idx & 15);
    long long s = idx >> 4;
    float m = (float)mask[s];
    long long b = s / HW, r = s % HW;
    float v = (c < 3) ? x[(b * 3 + c) * HW + r] * m : 0.0f;
    act[idx] = (__bf16)v;
    if (c == 0) mf[s] = m;
  }
}

// ---------------------------------------------------------------------------
// 3x3 same-pad conv as implicit GEMM. One wave = 16(cout) x 16(sites) tile.
// Branch-free B fragments: clamped address + AND-mask for border/partial-K.
// ---------------------------------------------------------------------------
__global__ __launch_bounds__(256)
void conv3x3_wmma(const __bf16* __restrict__ act, const __bf16* __restrict__ wp,
                  float* __restrict__ out, int B, int H, int W,
                  int cinS, int cin, int cout) {
  const int lane = threadIdx.x & 31;
  const int wave = threadIdx.x >> 5;
  const int n = lane & 15;
  const int half = lane >> 4;
  const int nkc = (cin + 31) >> 5;
  const int nfull = cin >> 5;
  const bool hasPartial = (cin & 31) != 0;       // wave-uniform
  const long long HW = (long long)H * W;
  const int nSiteTiles = (int)(((long long)B * HW) >> 4);
  int st = blockIdx.x * 8 + wave;
  if (st >= nSiteTiles) return;                  // wave-uniform
  const int mt = blockIdx.y;

  long long s = (long long)st * 16 + n;
  int b = (int)(s / HW);
  int r = (int)(s % HW);
  int y = r / W, x = r % W;

  // per-lane center pointer (always valid); per-tap offset is a SALU constant
  const __bf16* pc = act + ((long long)(b * H + y) * W + x) * cinS + (half << 4);
  const __bf16* wpm = wp + (((long long)mt * 9 * nkc) << 9) + (lane << 4);

  v8f acc = {};
  for (int t = 0; t < 9; ++t) {
    int dy = t / 3 - 1, dx = t % 3 - 1;
    int iy = y + dy, ix = x + dx;
    bool valid = ((unsigned)iy < (unsigned)H) && ((unsigned)ix < (unsigned)W);
    int offt = valid ? (dy * W + dx) * cinS : 0;          // one v_cndmask
    unsigned mskV = valid ? 0xffffffffu : 0u;
    const __bf16* ap = pc + offt;
    const __bf16* wpt = wpm + (((long long)t * nkc) << 9);
    for (int kc = 0; kc < nfull; ++kc) {
      bf16x16 Af = *(const bf16x16*)(wpt + ((long long)kc << 9));
      u32x8 braw = *(const u32x8*)(ap + (kc << 5));
      braw &= mskV;                                       // 8 dual-issue v_and
      bf16x16 Bf = __builtin_bit_cast(bf16x16, braw);
      acc = __builtin_amdgcn_wmma_f32_16x16x32_bf16(false, Af, false, Bf,
                                                    (short)0, acc, false, false);
    }
    if (hasPartial) {
      unsigned mskP = (valid && half == 0) ? 0xffffffffu : 0u;
      bf16x16 Af = *(const bf16x16*)(wpt + ((long long)nfull << 9));
      u32x8 braw = *(const u32x8*)(ap + (nfull << 5));
      braw &= mskP;
      bf16x16 Bf = __builtin_bit_cast(bf16x16, braw);
      acc = __builtin_amdgcn_wmma_f32_16x16x32_bf16(false, Af, false, Bf,
                                                    (short)0, acc, false, false);
    }
  }
  float* op = out + ((long long)(b * H + y) * W + x) * cout + mt * 16 + (half << 3);
  *(float4*)op       = make_float4(acc[0], acc[1], acc[2], acc[3]);
  *(float4*)(op + 4) = make_float4(acc[4], acc[5], acc[6], acc[7]);
}

// ---------------------------------------------------------------------------
// sparse-BN stats: [0..127]=sum(x*m), [128..255]=sum(x^2*m), [256]=count(m)
// ---------------------------------------------------------------------------
__global__ void zero_stats_kernel(float* __restrict__ s) {
  for (int i = threadIdx.x; i < 257; i += blockDim.x) s[i] = 0.0f;
}

__global__ void stats_kernel(const float* __restrict__ x, const float* __restrict__ mask,
                             float* __restrict__ stats, int C, long long nsites) {
  __shared__ float s1[128];
  __shared__ float s2[128];
  __shared__ float sc;
  if (threadIdx.x < C) { s1[threadIdx.x] = 0.0f; s2[threadIdx.x] = 0.0f; }
  if (threadIdx.x == 0) sc = 0.0f;
  __syncthreads();
  long long total = nsites * C;
  for (long long idx = (long long)blockIdx.x * blockDim.x + threadIdx.x; idx < total;
       idx += (long long)gridDim.x * blockDim.x) {
    int c = (int)(idx % C);
    long long s = idx / C;
    float m = mask[s];
    float v = x[idx] * m;               // x stride == C
    atomicAdd(&s1[c], v);
    atomicAdd(&s2[c], v * v);
    if (c == 0) atomicAdd(&sc, m);
  }
  __syncthreads();
  if (threadIdx.x < C) {
    atomicAdd(&stats[threadIdx.x], s1[threadIdx.x]);
    atomicAdd(&stats[128 + threadIdx.x], s2[threadIdx.x]);
  }
  if (threadIdx.x == 0) atomicAdd(&stats[256], sc);
}

// ---------------------------------------------------------------------------
// fused BN -> ReLU -> mask -> bf16 NHWC (stride C) store
// ---------------------------------------------------------------------------
__global__ void bn_relu_kernel(const float* __restrict__ x, const float* __restrict__ mask,
                               const float* __restrict__ gamma, const float* __restrict__ beta,
                               const float* __restrict__ stats, __bf16* __restrict__ out,
                               int C, long long nsites) {
  long long total = nsites * C;
  float cnt = fmaxf(stats[256], 1.0f);
  for (long long idx = (long long)blockIdx.x * blockDim.x + threadIdx.x; idx < total;
       idx += (long long)gridDim.x * blockDim.x) {
    int c = (int)(idx % C);
    long long s = idx / C;
    float mean = stats[c] / cnt;
    float var  = stats[128 + c] / cnt - mean * mean;
    float yv = gamma[c] * (x[idx] - mean) * rsqrtf(var + BN_EPS) + beta[c];
    yv = fmaxf(yv, 0.0f) * mask[s];
    out[idx] = (__bf16)yv;
  }
}

// ---------------------------------------------------------------------------
// masked 3x3/stride-2 max-pool on h (bf16 NHWC stride C) and mask
// ---------------------------------------------------------------------------
__global__ void pool_kernel(const __bf16* __restrict__ in, const float* __restrict__ mi,
                            __bf16* __restrict__ out, float* __restrict__ mo,
                            int B, int Hin, int Win, int Hout, int Wout, int C) {
  long long total = (long long)B * Hout * Wout * C;
  for (long long idx = (long long)blockIdx.x * blockDim.x + threadIdx.x; idx < total;
       idx += (long long)gridDim.x * blockDim.x) {
    int c = (int)(idx % C);
    long long s = idx / C;
    int ox = (int)(s % Wout);
    long long t = s / Wout;
    int oy = (int)(t % Hout);
    int b  = (int)(t / Hout);
    float hmax = NEGBIG, mmax = 0.0f;
    #pragma unroll
    for (int dy = 0; dy < 3; ++dy)
      #pragma unroll
      for (int dx = 0; dx < 3; ++dx) {
        int iy = 2 * oy + dy, ix = 2 * ox + dx;
        long long si = (long long)(b * Hin + iy) * Win + ix;
        float m = mi[si];
        float v = (float)in[si * C + c];
        mmax = fmaxf(mmax, m);
        hmax = fmaxf(hmax, (m > 0.0f) ? v : NEGBIG);
      }
    long long so = (long long)(b * Hout + oy) * Wout + ox;
    out[so * C + c] = (__bf16)((mmax > 0.0f) ? hmax : 0.0f);
    if (c == 0) mo[so] = mmax;
  }
}

__global__ void mask_pool_final(const float* __restrict__ m3, float* __restrict__ mo) {
  int b = blockIdx.x * blockDim.x + threadIdx.x;
  if (b < 512) {
    float mm = 0.0f;
    #pragma unroll
    for (int i = 0; i < 9; ++i) mm = fmaxf(mm, m3[b * 9 + i]);
    mo[b] = mm;
  }
}

// ---------------------------------------------------------------------------
// final strided conv 96->128 == GEMM  M=128, N=512, K=864 (contiguous patches)
// ---------------------------------------------------------------------------
__global__ __launch_bounds__(256)
void final_conv_wmma(const __bf16* __restrict__ act, const __bf16* __restrict__ wp,
                     float* __restrict__ out) {
  int wv = blockIdx.x * 8 + (threadIdx.x >> 5);
  if (wv >= 256) return;                  // 8 mtiles * 32 ntiles
  const int lane = threadIdx.x & 31, n = lane & 15, half = lane >> 4;
  const int mt = wv & 7, nt = wv >> 3;
  const int bcol = nt * 16 + n;
  const __bf16* bp = act + (long long)bcol * 864 + (half << 4);
  const __bf16* apk = wp + (((long long)mt * 27) << 9) + (lane << 4);
  v8f acc = {};
  for (int kc = 0; kc < 27; ++kc) {
    bf16x16 Af = *(const bf16x16*)(apk + ((long long)kc << 9));
    bf16x16 Bf = *(const bf16x16*)(bp + (kc << 5));
    acc = __builtin_amdgcn_wmma_f32_16x16x32_bf16(false, Af, false, Bf,
                                                  (short)0, acc, false, false);
  }
  float* op = out + (long long)bcol * 128 + mt * 16 + (half << 3);
  *(float4*)op       = make_float4(acc[0], acc[1], acc[2], acc[3]);
  *(float4*)(op + 4) = make_float4(acc[4], acc[5], acc[6], acc[7]);
}

// ---------------------------------------------------------------------------
// classifier: out(512,3755) = h(512,128) @ Wlin^T + blin
// ---------------------------------------------------------------------------
__global__ __launch_bounds__(256)
void linear_wmma(const __bf16* __restrict__ h, const __bf16* __restrict__ wp,
                 const float* __restrict__ bl, float* __restrict__ out) {
  const int MT = 235;
  int wv = blockIdx.x * 8 + (threadIdx.x >> 5);
  if (wv >= MT * 32) return;
  const int lane = threadIdx.x & 31, n = lane & 15, half = lane >> 4;
  const int mt = wv % MT, nt = wv / MT;
  const int bcol = nt * 16 + n;
  const __bf16* bp = h + (long long)bcol * 128 + (half << 4);
  const __bf16* apk = wp + (((long long)mt * 4) << 9) + (lane << 4);
  v8f acc = {};
  #pragma unroll
  for (int kc = 0; kc < 4; ++kc) {
    bf16x16 Af = *(const bf16x16*)(apk + ((long long)kc << 9));
    bf16x16 Bf = *(const bf16x16*)(bp + (kc << 5));
    acc = __builtin_amdgcn_wmma_f32_16x16x32_bf16(false, Af, false, Bf,
                                                  (short)0, acc, false, false);
  }
  #pragma unroll
  for (int rr = 0; rr < 8; ++rr) {
    int M = mt * 16 + (half << 3) + rr;
    if (M < 3755) out[(long long)bcol * 3755 + M] = acc[rr] + bl[M];
  }
}

// ---------------------------------------------------------------------------
// host orchestration
// ---------------------------------------------------------------------------
extern "C" void kernel_launch(void* const* d_in, const int* in_sizes, int n_in,
                              void* d_out, int out_size, void* d_ws, size_t ws_size,
                              hipStream_t stream) {
  (void)in_sizes; (void)n_in; (void)out_size; (void)ws_size;
  const float* x    = (const float*)d_in[0];
  const int*   mask = (const int*)d_in[1];
  const float* convw[10]; const float* gam[10]; const float* bet[10];
  for (int i = 0; i < 10; ++i) {
    convw[i] = (const float*)d_in[2 + i];
    gam[i]   = (const float*)d_in[12 + i];
    bet[i]   = (const float*)d_in[22 + i];
  }
  const float* wfin = (const float*)d_in[32];
  const float* gamf = (const float*)d_in[33];
  const float* betf = (const float*)d_in[34];
  const float* Wlin = (const float*)d_in[35];
  const float* blin = (const float*)d_in[36];
  float* outp = (float*)d_out;

  char* p = (char*)d_ws;
  auto carve = [&](size_t bytes) -> char* {
    char* q = p; p += (bytes + 255) & ~(size_t)255; return q;
  };
  const long long maxAct = 512LL * 63 * 63 * 16;               // elems (C<=16 at 63x63)
  __bf16* actA  = (__bf16*)carve((size_t)maxAct * 2);
  __bf16* actB  = (__bf16*)carve((size_t)maxAct * 2);
  float*  convf = (float*)carve((size_t)maxAct * 4);
  float*  maskA = (float*)carve((size_t)(512LL * 63 * 63) * 4);
  float*  maskB = (float*)carve((size_t)(512LL * 63 * 63) * 4);
  float*  stats = (float*)carve(257 * 4);
  __bf16* wpack = (__bf16*)carve((size_t)(235LL * 4 * 512 * 16) * 2);  // largest pack

  static const int Hs[5]     = {63, 31, 15, 7, 3};
  static const int cins[10]  = {3, 16, 16, 32, 32, 48, 48, 64, 64, 96};
  static const int couts[10] = {16, 16, 32, 32, 48, 48, 64, 64, 96, 96};

  __bf16* cur = actA; __bf16* oth = actB;
  float* mcur = maskA; float* moth = maskB;

  {
    long long tot = 512LL * 63 * 63 * 16;
    prep_kernel<<<(int)((tot + 255) / 256), 256, 0, stream>>>(x, mask, cur, mcur, 512, 63, 63);
  }

  int li = 0;
  int curS = 16;                                  // current activation channel stride
  for (int stage = 0; stage < 5; ++stage) {
    int H = Hs[stage];
    long long nsites = 512LL * H * H;
    for (int k = 0; k < 2; ++k) {
      int cin = cins[li], cout = couts[li];
      int nkc = (cin + 31) >> 5;
      long long packTot = ((long long)(cout >> 4) * 9 * nkc) << 9;
      pack_conv_kernel<<<(int)((packTot + 255) / 256), 256, 0, stream>>>(
          convw[li], wpack, cin, cout);
      int siteTiles = (int)(nsites / 16);
      dim3 grid((siteTiles + 7) / 8, cout / 16);
      conv3x3_wmma<<<grid, 256, 0, stream>>>(cur, wpack, convf, 512, H, H, curS, cin, cout);
      zero_stats_kernel<<<1, 256, 0, stream>>>(stats);
      stats_kernel<<<2048, 256, 0, stream>>>(convf, mcur, stats, cout, nsites);
      long long tot = nsites * cout;
      bn_relu_kernel<<<(int)((tot + 255) / 256), 256, 0, stream>>>(
          convf, mcur, gam[li], bet[li], stats, oth, cout, nsites);
      { __bf16* t = cur; cur = oth; oth = t; }
      curS = cout;
      ++li;
    }
    if (stage < 4) {
      int Ho = Hs[stage + 1];
      long long toto = 512LL * Ho * Ho * curS;
      long long blk = (toto + 255) / 256; if (blk > 32768) blk = 32768;
      pool_kernel<<<(int)blk, 256, 0, stream>>>(cur, mcur, oth, moth, 512, H, H, Ho, Ho, curS);
      { __bf16* t = cur; cur = oth; oth = t; }
      { float* t = mcur; mcur = moth; moth = t; }
    }
  }

  // final strided conv (GEMM M=128,N=512,K=864), final sparse BN, classifier
  {
    long long packTot = (8LL * 27) << 9;
    pack_final_kernel<<<(int)((packTot + 255) / 256), 256, 0, stream>>>(wfin, wpack);
    final_conv_wmma<<<32, 256, 0, stream>>>(cur, wpack, convf);
    mask_pool_final<<<2, 256, 0, stream>>>(mcur, moth);
    zero_stats_kernel<<<1, 256, 0, stream>>>(stats);
    stats_kernel<<<64, 256, 0, stream>>>(convf, moth, stats, 128, 512);
    bn_relu_kernel<<<(512 * 128 + 255) / 256, 256, 0, stream>>>(
        convf, moth, gamf, betf, stats, oth, 128, 512);
    long long packTotL = (235LL * 4) << 9;
    pack_linear_kernel<<<(int)((packTotL + 255) / 256), 256, 0, stream>>>(Wlin, wpack);
    linear_wmma<<<(235 * 32 + 7) / 8, 256, 0, stream>>>(oth, wpack, blin, outp);
  }
}
